// FDAF_67370857005493
// MI455X (gfx1250) — compile-verified, matrix operand order checked by reading, not run on hardware
//
#include <hip/hip_runtime.h>
#include <hip/hip_bf16.h>
#include <math.h>

typedef __attribute__((ext_vector_type(16))) _Float16 v16h;
typedef __attribute__((ext_vector_type(8)))  float    v8f;

#define N_IMG 4
#define C_IN  64
#define C2    128
#define HDIM  256
#define WDIM  256
#define HW    65536           // 256*256
#define PLANES (N_IMG * C2)   // 512
#define EPSV  1e-05f

// Branchless exact-GELU: 0.5*v*(1+erf(v/sqrt(2))) with A&S 7.1.26 erf
// (max abs err 2.5e-5) using the 1-instruction v_rcp_f32 / v_exp_f32 paths.
// Error is far below the f16 quantization applied to the result.
__device__ __forceinline__ float gelu_branchless(float v) {
    const float z  = v * 0.70710678118654752f;
    const float az = fabsf(z);
    const float t  = __builtin_amdgcn_rcpf(fmaf(0.3275911f, az, 1.0f));
    const float poly = t * (0.254829592f +
                       t * (-0.284496736f +
                       t * (1.421413741f +
                       t * (-1.453152027f +
                       t * 1.061405429f))));
    const float e = __expf(-az * az);
    const float erfv = copysignf(1.0f - poly * e, z);
    return 0.5f * v * (1.0f + erfv);
}

// ---------------------------------------------------------------------------
// Kernel 0: zero the stats accumulators (sum, sumsq). Must run every launch
// since we atomicAdd into them and the harness does not re-poison buffers.
// ---------------------------------------------------------------------------
__global__ void fdaf_zero_stats(float* __restrict__ stats /* 2*PLANES floats */) {
    int i = blockIdx.x * blockDim.x + threadIdx.x;
    if (i < 2 * PLANES) stats[i] = 0.0f;
}

// ---------------------------------------------------------------------------
// Kernel 1: depthwise 5x5 conv (pad 2) + bias over concat(x1,x2) -> y
// Also block-reduces sum / sumsq per (n,c2) plane into global atomics.
// grid: (HW/256, PLANES), block: 256
// ---------------------------------------------------------------------------
__global__ __launch_bounds__(256)
void fdaf_dwconv(const float* __restrict__ x1, const float* __restrict__ x2,
                 const float* __restrict__ dw_w, const float* __restrict__ dw_b,
                 float* __restrict__ y,
                 float* __restrict__ psum, float* __restrict__ psq) {
    const int plane = blockIdx.y;                  // n*C2 + c2
    const int n  = plane >> 7;
    const int c2 = plane & (C2 - 1);
    const int pix = blockIdx.x * 256 + threadIdx.x;
    const int h = pix >> 8;
    const int w = pix & (WDIM - 1);

    const float* src = (c2 < C_IN)
        ? x1 + ((size_t)n * C_IN + c2) * HW
        : x2 + ((size_t)n * C_IN + (c2 - C_IN)) * HW;
    const float* wgt = dw_w + c2 * 25;

    float acc = dw_b[c2];
#pragma unroll
    for (int dy = 0; dy < 5; ++dy) {
        int yy = h + dy - 2;
        if (yy < 0 || yy >= HDIM) continue;
#pragma unroll
        for (int dx = 0; dx < 5; ++dx) {
            int xx = w + dx - 2;
            if (xx < 0 || xx >= WDIM) continue;
            acc = fmaf(src[yy * WDIM + xx], wgt[dy * 5 + dx], acc);
        }
    }
    y[(size_t)plane * HW + pix] = acc;

    // block reduction of sum and sum-of-squares (wave32: 8 waves per block)
    __shared__ float s0[256];
    __shared__ float s1[256];
    s0[threadIdx.x] = acc;
    s1[threadIdx.x] = acc * acc;
    __syncthreads();
    for (int off = 128; off > 0; off >>= 1) {
        if (threadIdx.x < off) {
            s0[threadIdx.x] += s0[threadIdx.x + off];
            s1[threadIdx.x] += s1[threadIdx.x + off];
        }
        __syncthreads();
    }
    if (threadIdx.x == 0) {
        atomicAdd(&psum[plane], s0[0]);
        atomicAdd(&psq[plane],  s1[0]);
    }
}

// ---------------------------------------------------------------------------
// Kernel 2: finalize per-plane mean / rstd into an interleaved float2 table
// so the consumer needs a single b64 load per channel.  512 lanes total.
// ---------------------------------------------------------------------------
__global__ void fdaf_finalize_stats(const float* __restrict__ psum,
                                    const float* __restrict__ psq,
                                    float2* __restrict__ mnrs) {
    int p = blockIdx.x * blockDim.x + threadIdx.x;
    if (p < PLANES) {
        const float inv = 1.0f / (float)HW;        // exact: HW = 2^16
        float m = psum[p] * inv;
        float v = psq[p] * inv - m * m;
        mnrs[p] = make_float2(m, rsqrtf(v + EPSV));
    }
}

// ---------------------------------------------------------------------------
// Kernel 3: instance-norm + GELU + 1x1 conv (K=128 -> 4) via WMMA.
// One wave handles a 16-pixel x 16-outchan tile (4 real out channels,
// 12 zero-padded columns) with four v_wmma_f32_16x16x32_f16 along K.
//
// A (16x32 f16) ISA layout: lane holds M = lane&15; element i holds
//   K = 16*(i>>3) + 8*(lane>=16) + (i&7).
// B (32x16 f16) transpose-symmetric: lane holds N = lane&15, same K map.
// D (16x16 f32): element r -> M = r + 8*(lane>=16), N = lane&15.
//
// Fully branchless up to the WMMA ops: EXEC is all-1s structurally.
// grid: 2048 blocks x 256 threads (8 waves) = 16384 wave-tiles * 16 px.
// ---------------------------------------------------------------------------
__global__ __launch_bounds__(256)
void fdaf_pointwise_wmma(const float* __restrict__ y,
                         const float* __restrict__ pw_w,
                         const float2* __restrict__ mnrs,
                         float* __restrict__ flow) {
    const int lane = threadIdx.x & 31;
    const int wave = threadIdx.x >> 5;
    const int tile = blockIdx.x * 8 + wave;        // 0 .. 16383
    const int n    = tile >> 12;                   // 4096 tiles per image
    const int pix0 = (tile & 4095) << 4;           // 16 pixels per tile
    const int half = lane >> 4;                    // 0 or 1
    const int mloc = lane & 15;

    const float*  yimg = y    + (size_t)n * C2 * HW;
    const float2* mr   = mnrs + n * C2;
    const int pixA = pix0 + mloc;                  // pixel this lane feeds to A

    // ---- build B once (pixel-invariant), branchless: clamp row + keep-mask
    const int   wsel = (mloc < 4) ? mloc : 0;
    const float keep = (mloc < 4) ? 1.0f : 0.0f;
    const float* wrow = pw_w + wsel * C2;
    v16h bmat[4];
#pragma unroll
    for (int kt = 0; kt < 4; ++kt) {
        const int cb = kt * 32 + (half << 3);
#pragma unroll
        for (int i = 0; i < 16; ++i) {
            const int c = cb + ((i >> 3) << 4) + (i & 7);
            bmat[kt][i] = (_Float16)(keep * wrow[c]);
        }
    }

    v8f acc = {};
#pragma unroll
    for (int kt = 0; kt < 4; ++kt) {
        const int cb = kt * 32 + (half << 3);
        v16h a;
#pragma unroll
        for (int i = 0; i < 16; ++i) {
            const int c = cb + ((i >> 3) << 4) + (i & 7);
            const float2 ms = mr[c];
            float v = (yimg[(size_t)c * HW + pixA] - ms.x) * ms.y;
            a[i] = (_Float16)gelu_branchless(v);
        }
        acc = __builtin_amdgcn_wmma_f32_16x16x32_f16(
            /*neg_a=*/false, a, /*neg_b=*/false, bmat[kt],
            /*c_mod=*/(short)0, acc, /*reuse_a=*/false, /*reuse_b=*/false);
    }

    if (mloc < 4) {
        float* fl = flow + ((size_t)n * 4 + mloc) * HW;
#pragma unroll
        for (int r = 0; r < 8; ++r) {
            fl[pix0 + r + half * 8] = acc[r];
        }
    }
}

// ---------------------------------------------------------------------------
// Kernel 4: bilinear warp (align_corners=True, zero padding) + subtraction.
// blockIdx.y selects which output: 0 -> warp(x1,f1)-x2, 1 -> warp(x2,f2)-x1.
// All divisions replaced by reciprocal-constant multiplies.
// ---------------------------------------------------------------------------
__global__ __launch_bounds__(256)
void fdaf_warp(const float* __restrict__ x1, const float* __restrict__ x2,
               const float* __restrict__ flow, float* __restrict__ out) {
    const int which = blockIdx.y;                  // 0 or 1
    const int p = blockIdx.x * 256 + threadIdx.x;  // 0 .. N*HW-1
    const int n   = p >> 16;
    const int pix = p & (HW - 1);
    const int h = pix >> 8;
    const int w = pix & (WDIM - 1);

    const float fx = flow[((size_t)n * 4 + which * 2 + 0) * HW + pix];
    const float fy = flow[((size_t)n * 4 + which * 2 + 1) * HW + pix];

    // base grid in [-1,1] then +flow/{W,H}, then map to pixel coords
    const float gxn = fmaf((float)w, 2.0f / 255.0f, -1.0f) + fx * (1.0f / 256.0f);
    const float gyn = fmaf((float)h, 2.0f / 255.0f, -1.0f) + fy * (1.0f / 256.0f);
    const float gx = (gxn + 1.0f) * (0.5f * 255.0f);
    const float gy = (gyn + 1.0f) * (0.5f * 255.0f);

    const float x0f = floorf(gx), y0f = floorf(gy);
    const float wx1 = gx - x0f, wx0 = 1.0f - wx1;
    const float wy1 = gy - y0f, wy0 = 1.0f - wy1;
    const float x1f = x0f + 1.0f, y1f = y0f + 1.0f;

    const float mx0 = (x0f >= 0.0f && x0f <= 255.0f) ? 1.0f : 0.0f;
    const float mx1 = (x1f >= 0.0f && x1f <= 255.0f) ? 1.0f : 0.0f;
    const float my0 = (y0f >= 0.0f && y0f <= 255.0f) ? 1.0f : 0.0f;
    const float my1 = (y1f >= 0.0f && y1f <= 255.0f) ? 1.0f : 0.0f;

    const int xi0 = (int)fminf(fmaxf(x0f, 0.0f), 255.0f);
    const int xi1 = (int)fminf(fmaxf(x1f, 0.0f), 255.0f);
    const int yi0 = (int)fminf(fmaxf(y0f, 0.0f), 255.0f);
    const int yi1 = (int)fminf(fmaxf(y1f, 0.0f), 255.0f);

    const float w00 = wx0 * wy0 * mx0 * my0;
    const float w10 = wx1 * wy0 * mx1 * my0;
    const float w01 = wx0 * wy1 * mx0 * my1;
    const float w11 = wx1 * wy1 * mx1 * my1;

    const int i00 = yi0 * WDIM + xi0;
    const int i10 = yi0 * WDIM + xi1;
    const int i01 = yi1 * WDIM + xi0;
    const int i11 = yi1 * WDIM + xi1;

    const float* src = (which == 0) ? x1 : x2;
    const float* sub = (which == 0) ? x2 : x1;
    float* dst = out + (size_t)which * N_IMG * C_IN * HW;

    for (int c = 0; c < C_IN; ++c) {
        const float* sp = src + ((size_t)n * C_IN + c) * HW;
        float v = w00 * sp[i00] + w10 * sp[i10] + w01 * sp[i01] + w11 * sp[i11];
        const size_t o = ((size_t)n * C_IN + c) * HW + pix;
        dst[o] = v - sub[o];
    }
}

// ---------------------------------------------------------------------------
extern "C" void kernel_launch(void* const* d_in, const int* in_sizes, int n_in,
                              void* d_out, int out_size, void* d_ws, size_t ws_size,
                              hipStream_t stream) {
    const float* x1   = (const float*)d_in[0];
    const float* x2   = (const float*)d_in[1];
    const float* dw_w = (const float*)d_in[2];
    const float* dw_b = (const float*)d_in[3];
    const float* pw_w = (const float*)d_in[4];
    float* out = (float*)d_out;

    // workspace layout
    float* y    = (float*)d_ws;                    // N*C2*HW   = 33,554,432 f
    float* flow = y + (size_t)N_IMG * C2 * HW;     // N*4*HW    =  1,048,576 f
    float* psum = flow + (size_t)N_IMG * 4 * HW;   // PLANES
    float* psq  = psum + PLANES;
    float2* mnrs = (float2*)(psq + PLANES);        // PLANES float2 (8B aligned)

    // 0) zero stats accumulators (deterministic across graph replays)
    fdaf_zero_stats<<<dim3(4), dim3(256), 0, stream>>>(psum); // zeros psum+psq

    // 1) depthwise conv + bias + plane sum/sumsq
    fdaf_dwconv<<<dim3(HW / 256, PLANES), dim3(256), 0, stream>>>(
        x1, x2, dw_w, dw_b, y, psum, psq);

    // 2) mean / rstd (interleaved)
    fdaf_finalize_stats<<<dim3(2), dim3(256), 0, stream>>>(psum, psq, mnrs);

    // 3) norm + GELU + 1x1 conv via WMMA -> flow
    fdaf_pointwise_wmma<<<dim3(2048), dim3(256), 0, stream>>>(
        y, pw_w, mnrs, flow);

    // 4) bilinear warp + subtract -> out (both halves)
    fdaf_warp<<<dim3((N_IMG * HW) / 256, 2), dim3(256), 0, stream>>>(
        x1, x2, flow, out);
}
